// AdvancedRvNN_4939212391079
// MI455X (gfx1250) — matrix operand order repaired and checked by reading, not compiled, not run on hardware
//
#include <hip/hip_runtime.h>
#include <math.h>

typedef __attribute__((ext_vector_type(16))) _Float16 v16h;
typedef __attribute__((ext_vector_type(8)))  float    v8f;

#define N_NODES 65536
#define N_GRAPHS 512
#define T_GRAPH 128
#define N_EDGES 524288
#define E_TOT   (N_EDGES + N_NODES)   // self-loops appended
#define D_IN 768
#define HID 128
#define HEADS 4
#define N_BLOCKS 3
#define N_CLASSES 4

// ---------------------------------------------------------------------------
// Weight transpose + f32->f16: Bt16[n*K + k] = (f16) B[k*N + n]
// ---------------------------------------------------------------------------
__global__ void transpose_f16_kernel(const float* __restrict__ B, _Float16* __restrict__ Bt,
                                     int K, int N)
{
    int idx = blockIdx.x * blockDim.x + threadIdx.x;
    if (idx >= K * N) return;
    int n = idx / K, k = idx - n * K;
    Bt[idx] = (_Float16)B[(size_t)k * N + n];
}

// Straight f32->f16 convert (for weights already stored [N x K])
__global__ void convert_f16_kernel(const float* __restrict__ W, _Float16* __restrict__ Wt,
                                   int count)
{
    int idx = blockIdx.x * blockDim.x + threadIdx.x;
    if (idx >= count) return;
    Wt[idx] = (_Float16)W[idx];
}

// ---------------------------------------------------------------------------
// f16-WMMA GEMM: C[MxN] = epilogue(A[MxK](f32) * Bt^T), Bt is f16 [N x K].
// One wave computes a 16x64 tile (4 accumulators); A fragment reused 4x.
// B fragment = one contiguous 32-byte v16h load. All loads of a k-step are
// issued before any conversion/WMMA so they batch into one clause.
// EPI: 0 none, 1 +bias, 2 gelu(bn(+bias)).
// ---------------------------------------------------------------------------
template <int EPI>
__global__ void __launch_bounds__(256) gemm_bt_wmma(
    const float* __restrict__ A, const _Float16* __restrict__ Bt, float* __restrict__ C,
    int M, int N, int K,
    const float* __restrict__ bias,
    const float* __restrict__ bn_g, const float* __restrict__ bn_b,
    const float* __restrict__ bn_m, const float* __restrict__ bn_v)
{
    long wave   = ((long)blockIdx.x * blockDim.x + threadIdx.x) >> 5;
    int  tilesN = N >> 6;                       // 64-wide column tiles
    long tiles  = (long)(M >> 4) * tilesN;
    if (wave >= tiles) return;
    int lane = threadIdx.x & 31;
    int tm = (int)(wave / tilesN) << 4;
    int tn = (int)(wave % tilesN) << 6;

    int hi  = lane >> 4;          // 0: lanes 0-15, 1: lanes 16-31
    int l15 = lane & 15;
    int m   = tm + l15;           // A row held by this lane
    int akb = hi * 16;            // A: K offset 0/16
    int bkb = hi * 16;            // B: K offset 0/16

    v8f acc0 = {0.f,0.f,0.f,0.f,0.f,0.f,0.f,0.f};
    v8f acc1 = acc0, acc2 = acc0, acc3 = acc0;

    const float*    arow = A + (size_t)m * K + akb;
    const _Float16* bp0  = Bt + (size_t)(tn +       l15) * K + bkb;
    const _Float16* bp1  = Bt + (size_t)(tn + 16  + l15) * K + bkb;
    const _Float16* bp2  = Bt + (size_t)(tn + 32  + l15) * K + bkb;
    const _Float16* bp3  = Bt + (size_t)(tn + 48  + l15) * K + bkb;

    for (int k0 = 0; k0 < K; k0 += 32) {
        // ---- issue all loads first (independent destinations) ----
        float4 x0 = ((const float4*)(arow + k0))[0];
        float4 x1 = ((const float4*)(arow + k0))[1];
        float4 y0 = ((const float4*)(arow + k0 + 8))[0];
        float4 y1 = ((const float4*)(arow + k0 + 8))[1];
        v16h b0 = *(const v16h*)(bp0 + k0);
        v16h b1 = *(const v16h*)(bp1 + k0);
        v16h b2 = *(const v16h*)(bp2 + k0);
        v16h b3 = *(const v16h*)(bp3 + k0);
        // ---- convert A fragment ----
        v16h af;
        af[0]=(_Float16)x0.x; af[1]=(_Float16)x0.y; af[2]=(_Float16)x0.z; af[3]=(_Float16)x0.w;
        af[4]=(_Float16)x1.x; af[5]=(_Float16)x1.y; af[6]=(_Float16)x1.z; af[7]=(_Float16)x1.w;
        af[8]=(_Float16)y0.x; af[9]=(_Float16)y0.y; af[10]=(_Float16)y0.z; af[11]=(_Float16)y0.w;
        af[12]=(_Float16)y1.x; af[13]=(_Float16)y1.y; af[14]=(_Float16)y1.z; af[15]=(_Float16)y1.w;
        // ---- 4 back-to-back WMMAs ----
        acc0 = __builtin_amdgcn_wmma_f32_16x16x32_f16(false, af, false, b0, (short)0, acc0, false, false);
        acc1 = __builtin_amdgcn_wmma_f32_16x16x32_f16(false, af, false, b1, (short)0, acc1, false, false);
        acc2 = __builtin_amdgcn_wmma_f32_16x16x32_f16(false, af, false, b2, (short)0, acc2, false, false);
        acc3 = __builtin_amdgcn_wmma_f32_16x16x32_f16(false, af, false, b3, (short)0, acc3, false, false);
    }

    // ---- store: VGPR r -> row tm + r + 8*hi, col (tile) + l15 ----
    int row0 = tm + hi * 8;
    #pragma unroll
    for (int u = 0; u < 4; ++u) {
        int col = tn + u * 16 + l15;
        float scale = 0.f, shift = 0.f;
        if (EPI == 2) {
            scale = rsqrtf(bn_v[col] + 1e-5f) * bn_g[col];
            shift = bn_b[col] - bn_m[col] * scale;
        }
        #pragma unroll
        for (int r = 0; r < 8; ++r) {
            float v = (u == 0) ? acc0[r] : (u == 1) ? acc1[r] : (u == 2) ? acc2[r] : acc3[r];
            if (EPI >= 1) v += bias[col];
            if (EPI == 2) {
                v = v * scale + shift;
                v = 0.5f * v * (1.0f + erff(v * 0.70710678118654752f)); // exact GELU
            }
            C[(size_t)(row0 + r) * N + col] = v;
        }
    }
}

// ---------------------------------------------------------------------------
// Ordered-float encoding for atomicMax on uint
// ---------------------------------------------------------------------------
__device__ __forceinline__ unsigned int fenc(float f) {
    unsigned int u = __float_as_uint(f);
    return (u & 0x80000000u) ? ~u : (u | 0x80000000u);
}
__device__ __forceinline__ float fdec(unsigned int u) {
    return (u & 0x80000000u) ? __uint_as_float(u & 0x7FFFFFFFu) : __uint_as_float(~u);
}

// ---------------------------------------------------------------------------
// Attention pass 1: per-edge logits (leaky_relu(xl[s]+xr[d]) . a) + segment max
// ---------------------------------------------------------------------------
__global__ void att_logits_max(const int* __restrict__ ei,
                               const float* __restrict__ xl, const float* __restrict__ xr,
                               const float* __restrict__ avec,
                               float* __restrict__ logits, unsigned int* __restrict__ maxbuf)
{
    int wave = (int)(((long)blockIdx.x * blockDim.x + threadIdx.x) >> 5);
    int lane = threadIdx.x & 31;
    if (wave >= E_TOT) return;
    int s, d;
    if (wave < N_EDGES) { s = ei[wave]; d = ei[N_EDGES + wave]; }
    else                { s = wave - N_EDGES; d = s; }
    const float* xls = xl + (size_t)s * (HEADS * HID);
    const float* xrd = xr + (size_t)d * (HEADS * HID);
    #pragma unroll
    for (int h = 0; h < HEADS; ++h) {
        float p = 0.f;
        #pragma unroll
        for (int j = 0; j < 4; ++j) {
            int c = lane + j * 32;
            float z = xls[h * HID + c] + xrd[h * HID + c];
            z = (z > 0.f) ? z : 0.2f * z;                 // leaky_relu(0.2)
            p += z * avec[h * HID + c];
        }
        #pragma unroll
        for (int off = 16; off; off >>= 1) p += __shfl_xor(p, off);
        if (lane == 0) {
            logits[(size_t)wave * HEADS + h] = p;
            atomicMax(maxbuf + (size_t)d * HEADS + h, fenc(p));
        }
    }
}

// ---------------------------------------------------------------------------
// Attention pass 2: ex = exp(logit - max[dst]); denom[dst] += ex
// ---------------------------------------------------------------------------
__global__ void att_expsum(const int* __restrict__ ei,
                           const float* __restrict__ logits,
                           const unsigned int* __restrict__ maxbuf,
                           float* __restrict__ ex, float* __restrict__ denom)
{
    int idx = blockIdx.x * blockDim.x + threadIdx.x;
    if (idx >= E_TOT * HEADS) return;
    int e = idx >> 2, h = idx & 3;
    int d = (e < N_EDGES) ? ei[N_EDGES + e] : (e - N_EDGES);
    float mx = fdec(maxbuf[(size_t)d * HEADS + h]);
    float v = expf(logits[idx] - mx);
    ex[idx] = v;
    atomicAdd(denom + (size_t)d * HEADS + h, v);
}

// ---------------------------------------------------------------------------
// Attention pass 3: out[dst,c] += (1/H) * sum_h alpha_h * xl[src,h,c]
// ---------------------------------------------------------------------------
__global__ void att_aggregate(const int* __restrict__ ei,
                              const float* __restrict__ xl,
                              const float* __restrict__ ex, const float* __restrict__ denom,
                              float* __restrict__ outb)
{
    int wave = (int)(((long)blockIdx.x * blockDim.x + threadIdx.x) >> 5);
    int lane = threadIdx.x & 31;
    if (wave >= E_TOT) return;
    int s, d;
    if (wave < N_EDGES) { s = ei[wave]; d = ei[N_EDGES + wave]; }
    else                { s = wave - N_EDGES; d = s; }
    float al[HEADS];
    #pragma unroll
    for (int h = 0; h < HEADS; ++h)
        al[h] = ex[(size_t)wave * HEADS + h] / denom[(size_t)d * HEADS + h];
    const float* xls = xl + (size_t)s * (HEADS * HID);
    #pragma unroll
    for (int j = 0; j < 4; ++j) {
        int c = lane + j * 32;
        float v = 0.25f * (al[0] * xls[c] + al[1] * xls[HID + c] +
                           al[2] * xls[2 * HID + c] + al[3] * xls[3 * HID + c]);
        atomicAdd(outb + (size_t)d * HID + c, v);
    }
}

// ---------------------------------------------------------------------------
// GraphNorm + ELU + residual (in-place h update). One block per graph.
// ---------------------------------------------------------------------------
__global__ void __launch_bounds__(HID) graphnorm_elu_res(
    const float* __restrict__ outb, const float* __restrict__ bias,
    const float* __restrict__ gng, const float* __restrict__ gnb,
    const float* __restrict__ gna, float* __restrict__ h)
{
    int g = blockIdx.x, c = threadIdx.x;
    size_t base = (size_t)g * T_GRAPH * HID + c;
    float b = bias[c];
    float s = 0.f;
    for (int t = 0; t < T_GRAPH; ++t) s += outb[base + (size_t)t * HID];
    float mean = s * (1.f / T_GRAPH) + b;
    float am = gna[c] * mean;
    float var = 0.f;
    for (int t = 0; t < T_GRAPH; ++t) {
        float cent = outb[base + (size_t)t * HID] + b - am;
        var += cent * cent;
    }
    float inv = rsqrtf(var * (1.f / T_GRAPH) + 1e-5f) * gng[c];
    float bb = gnb[c];
    for (int t = 0; t < T_GRAPH; ++t) {
        float cent = outb[base + (size_t)t * HID] + b - am;
        float nv = cent * inv + bb;
        float e = (nv > 0.f) ? nv : (expf(nv) - 1.f);   // elu
        h[base + (size_t)t * HID] += e;                  // + residual
    }
}

// ---------------------------------------------------------------------------
// GRU scan: one block per graph, 384 threads. Whh staged transposed in LDS
// (~194KB of the WGP's 320KB). Outputs time-mean of hidden states.
// ---------------------------------------------------------------------------
__global__ void __launch_bounds__(384) gru_scan(
    const float* __restrict__ gi, const float* __restrict__ Whh,
    const float* __restrict__ bhh, float* __restrict__ rnn_mean, int reverse)
{
    __shared__ float sWT[HID * 3 * HID];   // sWT[k*384 + j] = Whh[j*128 + k]
    __shared__ float sh[HID];
    __shared__ float sgh[3 * HID];
    int g = blockIdx.x, tid = threadIdx.x;
    for (int idx = tid; idx < HID * 3 * HID; idx += 384) {
        int k = idx / 384, j = idx % 384;
        sWT[idx] = Whh[(size_t)j * HID + k];
    }
    if (tid < HID) sh[tid] = 0.f;
    float bj = bhh[tid];
    float accum = 0.f;
    __syncthreads();
    for (int t = 0; t < T_GRAPH; ++t) {
        int node = (g << 7) + (reverse ? (T_GRAPH - 1 - t) : t);
        float a = bj;
        const float* w = sWT + tid;
        #pragma unroll 8
        for (int k = 0; k < HID; ++k) a += sh[k] * w[(size_t)k * 384];
        sgh[tid] = a;
        __syncthreads();
        if (tid < HID) {
            const float* gp = gi + (size_t)node * (3 * HID);
            float r  = 1.f / (1.f + expf(-(gp[tid]         + sgh[tid])));
            float z  = 1.f / (1.f + expf(-(gp[HID + tid]   + sgh[HID + tid])));
            float nn = tanhf(gp[2 * HID + tid] + r * sgh[2 * HID + tid]);
            float hnew = (1.f - z) * nn + z * sh[tid];
            accum += hnew;
            sh[tid] = hnew;
        }
        __syncthreads();
    }
    if (tid < HID) rnn_mean[(size_t)g * HID + tid] = accum * (1.f / T_GRAPH);
}

// ---------------------------------------------------------------------------
// Readout: per-graph mean/max/sum/std over nodes + RNN means -> combined[G,768]
// ---------------------------------------------------------------------------
__global__ void __launch_bounds__(HID) readout_kernel(
    const float* __restrict__ h, const float* __restrict__ rf,
    const float* __restrict__ rb, float* __restrict__ comb)
{
    int g = blockIdx.x, c = threadIdx.x;
    size_t base = (size_t)g * T_GRAPH * HID + c;
    float s = 0.f, mx = -3.402823466e38f;
    for (int t = 0; t < T_GRAPH; ++t) {
        float v = h[base + (size_t)t * HID];
        s += v; mx = fmaxf(mx, v);
    }
    float mean = s * (1.f / T_GRAPH);
    float var = 0.f;
    for (int t = 0; t < T_GRAPH; ++t) {
        float v = h[base + (size_t)t * HID] - mean;
        var += v * v;
    }
    float sd = sqrtf(var * (1.f / T_GRAPH) + 1e-6f);
    float* cg = comb + (size_t)g * (6 * HID);
    cg[c] = mean; cg[HID + c] = mx; cg[2 * HID + c] = s; cg[3 * HID + c] = sd;
    cg[4 * HID + c] = rf[(size_t)g * HID + c];
    cg[5 * HID + c] = rb[(size_t)g * HID + c];
}

// ---------------------------------------------------------------------------
// Final tiny classifier: out[g,c] = hid[g,:] . W2[:,c] + b2[c]
// ---------------------------------------------------------------------------
__global__ void cls2_kernel(const float* __restrict__ hid, const float* __restrict__ W2,
                            const float* __restrict__ b2, float* __restrict__ out)
{
    int idx = blockIdx.x * blockDim.x + threadIdx.x;
    if (idx >= N_GRAPHS * N_CLASSES) return;
    int g = idx >> 2, c = idx & 3;
    float s = b2[c];
    const float* hp = hid + (size_t)g * HID;
    #pragma unroll 8
    for (int k = 0; k < HID; ++k) s += hp[k] * W2[k * N_CLASSES + c];
    out[idx] = s;
}

// ---------------------------------------------------------------------------
template <int EPI>
static inline void launch_gemm(const float* A, const _Float16* Bt, float* C,
                               int M, int N, int K,
                               const float* bias, const float* g, const float* b,
                               const float* m, const float* v, hipStream_t s)
{
    long tiles = (long)(M / 16) * (N / 64);
    long total = tiles * 32;
    int block = 256;
    int grid = (int)((total + block - 1) / block);
    gemm_bt_wmma<EPI><<<grid, block, 0, s>>>(A, Bt, C, M, N, K, bias, g, b, m, v);
}

extern "C" void kernel_launch(void* const* d_in, const int* in_sizes, int n_in,
                              void* d_out, int out_size, void* d_ws, size_t ws_size,
                              hipStream_t stream) {
    (void)in_sizes; (void)n_in; (void)out_size; (void)ws_size;
    const float* x       = (const float*)d_in[0];
    const int*   ei      = (const int*)  d_in[1];
    // d_in[2] = batch (implicit: node/128)
    const float* proj_W  = (const float*)d_in[3];
    const float* proj_b  = (const float*)d_in[4];
    const float* bn0_g   = (const float*)d_in[5];
    const float* bn0_b   = (const float*)d_in[6];
    const float* bn0_m   = (const float*)d_in[7];
    const float* bn0_v   = (const float*)d_in[8];
    const float* att_Wl  = (const float*)d_in[9];
    const float* att_Wr  = (const float*)d_in[10];
    const float* att_a   = (const float*)d_in[11];
    const float* att_bias= (const float*)d_in[12];
    const float* gn_g    = (const float*)d_in[13];
    const float* gn_b    = (const float*)d_in[14];
    const float* gn_a    = (const float*)d_in[15];
    const float* Wih_f   = (const float*)d_in[16];
    const float* Whh_f   = (const float*)d_in[17];
    const float* bih_f   = (const float*)d_in[18];
    const float* bhh_f   = (const float*)d_in[19];
    const float* Wih_b   = (const float*)d_in[20];
    const float* Whh_b   = (const float*)d_in[21];
    const float* bih_b   = (const float*)d_in[22];
    const float* bhh_b   = (const float*)d_in[23];
    const float* cls_W1  = (const float*)d_in[24];
    const float* cls_b1  = (const float*)d_in[25];
    const float* bn1_g   = (const float*)d_in[26];
    const float* bn1_b   = (const float*)d_in[27];
    const float* bn1_m   = (const float*)d_in[28];
    const float* bn1_v   = (const float*)d_in[29];
    const float* cls_W2  = (const float*)d_in[30];
    const float* cls_b2  = (const float*)d_in[31];
    float* out = (float*)d_out;

    // ---- workspace layout (floats) ----
    float* ws = (float*)d_ws;
    size_t o = 0;
    float* h      = ws + o; o += (size_t)N_NODES * HID;          // 8.4M
    float* xl     = ws + o; o += (size_t)N_NODES * HEADS * HID;  // 33.6M (also gi_f)
    float* xr     = ws + o; o += (size_t)N_NODES * HEADS * HID;  // 33.6M (also gi_b)
    float* outb   = ws + o; o += (size_t)N_NODES * HID;          // 8.4M
    float* logits = ws + o; o += (size_t)E_TOT * HEADS;
    float* exbuf  = ws + o; o += (size_t)E_TOT * HEADS;
    unsigned int* maxbuf = (unsigned int*)(ws + o); o += (size_t)N_NODES * HEADS;
    float* denom  = ws + o; o += (size_t)N_NODES * HEADS;
    float* rnn_f  = ws + o; o += (size_t)N_GRAPHS * HID;
    float* rnn_b  = ws + o; o += (size_t)N_GRAPHS * HID;
    float* comb   = ws + o; o += (size_t)N_GRAPHS * 6 * HID;
    float* hid    = ws + o; o += (size_t)N_GRAPHS * HID;
    // f16 weight buffers (Bt[N x K] form); carve from float region (64B aligned)
    _Float16* f16base = (_Float16*)(ws + o);
    size_t ho = 0;
    _Float16* projWt = f16base + ho; ho += (size_t)HID * D_IN;                   // 128x768
    _Float16* Wlt    = f16base + ho; ho += (size_t)N_BLOCKS * HEADS * HID * HID; // 3x(512x128)
    _Float16* Wrt    = f16base + ho; ho += (size_t)N_BLOCKS * HEADS * HID * HID;
    _Float16* W1t    = f16base + ho; ho += (size_t)HID * 6 * HID;                // 128x768
    _Float16* Wih16f = f16base + ho; ho += (size_t)3 * HID * HID;                // 384x128
    _Float16* Wih16b = f16base + ho; ho += (size_t)3 * HID * HID;

    // 0) transpose/convert weights to f16 once (per launch; deterministic)
    transpose_f16_kernel<<<(D_IN * HID + 255) / 256, 256, 0, stream>>>(proj_W, projWt, D_IN, HID);
    for (int i = 0; i < N_BLOCKS; ++i) {
        transpose_f16_kernel<<<(HID * HEADS * HID + 255) / 256, 256, 0, stream>>>(
            att_Wl + (size_t)i * HID * HEADS * HID, Wlt + (size_t)i * HID * HEADS * HID,
            HID, HEADS * HID);
        transpose_f16_kernel<<<(HID * HEADS * HID + 255) / 256, 256, 0, stream>>>(
            att_Wr + (size_t)i * HID * HEADS * HID, Wrt + (size_t)i * HID * HEADS * HID,
            HID, HEADS * HID);
    }
    transpose_f16_kernel<<<(6 * HID * HID + 255) / 256, 256, 0, stream>>>(cls_W1, W1t, 6 * HID, HID);
    convert_f16_kernel<<<(3 * HID * HID + 255) / 256, 256, 0, stream>>>(Wih_f, Wih16f, 3 * HID * HID);
    convert_f16_kernel<<<(3 * HID * HID + 255) / 256, 256, 0, stream>>>(Wih_b, Wih16b, 3 * HID * HID);

    // 1) feature projection: h = gelu(bn0(x @ proj_W + proj_b))
    launch_gemm<2>(x, projWt, h, N_NODES, HID, D_IN,
                   proj_b, bn0_g, bn0_b, bn0_m, bn0_v, stream);

    // 2) three GATv2 + GraphNorm residual blocks
    const int edgeWaveGrid = (E_TOT * 32 + 255) / 256;
    for (int i = 0; i < N_BLOCKS; ++i) {
        const _Float16* WltI = Wlt + (size_t)i * HID * HEADS * HID;
        const _Float16* WrtI = Wrt + (size_t)i * HID * HEADS * HID;
        launch_gemm<0>(h, WltI, xl, N_NODES, HEADS * HID, HID, 0, 0, 0, 0, 0, stream);
        launch_gemm<0>(h, WrtI, xr, N_NODES, HEADS * HID, HID, 0, 0, 0, 0, 0, stream);
        hipMemsetAsync(maxbuf, 0, (size_t)N_NODES * HEADS * 4, stream);
        hipMemsetAsync(denom,  0, (size_t)N_NODES * HEADS * 4, stream);
        hipMemsetAsync(outb,   0, (size_t)N_NODES * HID * 4, stream);
        att_logits_max<<<edgeWaveGrid, 256, 0, stream>>>(
            ei, xl, xr, att_a + (size_t)i * HEADS * HID, logits, maxbuf);
        att_expsum<<<(E_TOT * HEADS + 255) / 256, 256, 0, stream>>>(
            ei, logits, maxbuf, exbuf, denom);
        att_aggregate<<<edgeWaveGrid, 256, 0, stream>>>(ei, xl, exbuf, denom, outb);
        graphnorm_elu_res<<<N_GRAPHS, HID, 0, stream>>>(
            outb, att_bias + (size_t)i * HID,
            gn_g + (size_t)i * HID, gn_b + (size_t)i * HID, gn_a + (size_t)i * HID, h);
    }

    // 3) GRU input transforms: gi = h @ Wih^T + bih
    float* gi_f = xl;
    float* gi_b = xr;
    launch_gemm<1>(h, Wih16f, gi_f, N_NODES, 3 * HID, HID, bih_f, 0, 0, 0, 0, stream);
    launch_gemm<1>(h, Wih16b, gi_b, N_NODES, 3 * HID, HID, bih_b, 0, 0, 0, 0, stream);

    // 4) bidirectional GRU scans (output = time-mean of hidden states)
    gru_scan<<<N_GRAPHS, 384, 0, stream>>>(gi_f, Whh_f, bhh_f, rnn_f, 0);
    gru_scan<<<N_GRAPHS, 384, 0, stream>>>(gi_b, Whh_b, bhh_b, rnn_b, 1);

    // 5) readout -> combined [G, 6H]
    readout_kernel<<<N_GRAPHS, HID, 0, stream>>>(h, rnn_f, rnn_b, comb);

    // 6) classifier head
    launch_gemm<2>(comb, W1t, hid, N_GRAPHS, HID, 6 * HID,
                   cls_b1, bn1_g, bn1_b, bn1_m, bn1_v, stream);
    cls2_kernel<<<(N_GRAPHS * N_CLASSES + 255) / 256, 256, 0, stream>>>(
        hid, cls_W2, cls_b2, out);
}
